// TCDiscriminator_46213848105433
// MI455X (gfx1250) — matrix operand order repaired and checked by reading, not compiled, not run on hardware
//
#include <hip/hip_runtime.h>
#include <hip/hip_bf16.h>

// ---------------------------------------------------------------------------
// TCDiscriminator on MI455X (gfx1250, wave32, WMMA).
// conv(width==E) == GEMM with overlapping-row A  ->  v_wmma_f32_16x16x32_f16
// v3: register-blocked (8 M-tile accumulators, B reused 8x per K-step)
//     + explicit 1-deep software pipeline on A/B fragments so each WMMA
//     waits with the next fragment loads still in flight.
// ---------------------------------------------------------------------------

typedef __attribute__((ext_vector_type(16))) _Float16 v16h;
typedef __attribute__((ext_vector_type(8)))  _Float16 v8h;
typedef __attribute__((ext_vector_type(8)))  float    v8f;

#define BB    2048
#define SS    128
#define EE    80
#define SP    160          // padded time rows per batch (safe for f<=20, Kpad<=1600)
#define FTOT  1720
#define NBR   12

// Highway GEMM geometry
#define HW_KT   54         // ceil(1720/32)
#define HW_NT   108        // ceil(1720/16)
#define HW_KPAD 1728

// ---- per-branch metadata (device) -----------------------------------------
__constant__ int c_n[NBR]      = {100,200,200,200,200,100,100,100,100,100,160,160};
__constant__ int c_ktiles[NBR] = {3,5,8,10,13,15,18,20,23,25,38,50};
__constant__ int c_cumnt[NBR]  = {0,7,20,33,46,59,66,73,80,87,94,104};   // 114 total
__constant__ int c_woff[NBR]   = {0,10752,44032,97280,163840,250368,
                                  304128,368640,440320,522752,612352,806912};
__constant__ int c_coloff[NBR] = {0,100,300,500,700,900,1000,1100,1200,1300,1400,1560};
__constant__ int c_sout[NBR]   = {128,127,126,125,124,123,122,121,120,119,114,109};

// ---- WMMA helpers ---------------------------------------------------------
__device__ __forceinline__ v8f wmma_f16(v16h a, v16h b, v8f c) {
  return __builtin_amdgcn_wmma_f32_16x16x32_f16(
      /*neg_a=*/false, a, /*neg_b=*/false, b,
      /*c_mod=*/(short)0, c, /*reuse_a=*/false, /*reuse_b=*/false);
}

// 16-bit A-fragment 16x32 (ISA 7.12.2): lanes 0-15 hold K 0..7 & 16..23,
// lanes 16-31 hold K 8..15 & 24..31, row M = lane&15.
__device__ __forceinline__ v16h load_a_frag(const _Float16* __restrict__ A,
                                            int lda, int mbase, int kbase, int lane) {
  int row = mbase + (lane & 15);
  int ko  = kbase + ((lane >> 4) << 3);
  const _Float16* p = A + (size_t)row * lda + ko;
  v8h lo = *(const v8h*)(p);
  v8h hi = *(const v8h*)(p + 16);
  return __builtin_shufflevector(lo, hi, 0,1,2,3,4,5,6,7,8,9,10,11,12,13,14,15);
}

// ---- kernel 1: embedding gather -> f16 [B, SP, E], zero time-pad ----------
__global__ void embed_gather(const int* __restrict__ xt, const int* __restrict__ xs,
                             const float* __restrict__ temb, const float* __restrict__ semb,
                             _Float16* __restrict__ emb) {
  size_t idx = (size_t)blockIdx.x * blockDim.x + threadIdx.x;
  if (idx >= (size_t)BB * SP * EE) return;
  int e = (int)(idx % EE);
  size_t bs = idx / EE;
  int s = (int)(bs % SP);
  int b = (int)(bs / SP);
  float v = 0.f;
  if (s < SS) {
    if (e < 16) v = temb[(size_t)xt[(size_t)b * SS + s] * 16 + e];
    else        v = semb[(size_t)xs[(size_t)b * SS + s] * 64 + (e - 16)];
  }
  emb[idx] = (_Float16)v;
}

// ---- kernel 2: pack [K,N] f32 row-major into wave32 B-fragment tiles ------
// Tile order: nt-major, kt-minor; within a 512-half tile: lane*16 + j, where
// col = nt*16 + (lane&15), k = kt*32 + ((lane>=16)?8:0) + (j&7) + ((j>=8)?16:0). Zero-padded.
__global__ void pack_b_f16(const float* __restrict__ src, _Float16* __restrict__ dst,
                           int K, int N, int ktiles, int ntiles) {
  int idx = blockIdx.x * blockDim.x + threadIdx.x;
  int total = ntiles * ktiles * 512;
  if (idx >= total) return;
  int tile = idx >> 9;
  int r    = idx & 511;
  int lane = r >> 4;
  int j    = r & 15;
  int nt = tile / ktiles;
  int kt = tile - nt * ktiles;
  int k   = kt * 32 + ((lane >> 4) << 3) + (j & 7) + ((j >> 3) << 4);
  int col = nt * 16 + (lane & 15);
  float v = (k < K && col < N) ? src[(size_t)k * N + col] : 0.f;
  dst[idx] = (_Float16)v;
}

__global__ void copy_bias(const float* __restrict__ src, float* __restrict__ dst, int n) {
  int i = blockIdx.x * blockDim.x + threadIdx.x;
  if (i < n) dst[i] = src[i];
}

// ---- kernel 3: conv-as-GEMM + bias + ReLU + time max-pool -----------------
// grid = (B, 114); 1 wave per (batch, branch-N-tile). All 8 time M-tiles held
// as live accumulators; B loaded once per K-step; A/B fragments software-
// pipelined one step ahead so WMMAs never wait on an empty load queue.
__global__ void __launch_bounds__(32)
conv_gemm_pool(const _Float16* __restrict__ emb, const _Float16* __restrict__ wpack,
               const float* __restrict__ biasCat, float* __restrict__ pred) {
  int b  = blockIdx.x;
  int ti = blockIdx.y;
  int br = 0;
  while (br + 1 < NBR && ti >= c_cumnt[br + 1]) ++br;
  int nt = ti - c_cumnt[br];

  int lane = threadIdx.x;
  const _Float16* Ab = emb + (size_t)b * SP * EE;
  int ktiles = c_ktiles[br];
  const _Float16* Bt = wpack + (size_t)c_woff[br] + (size_t)nt * ktiles * 512;
  int sout = c_sout[br];

  v8f acc[8];
  #pragma unroll
  for (int m = 0; m < 8; ++m) acc[m] = (v8f){};

  v16h bcur = *(const v16h*)(Bt + lane * 16);
  v16h acur = load_a_frag(Ab, EE, 0, 0, lane);

  for (int kt = 0; kt < ktiles; ++kt) {
    int ktn = (kt + 1 < ktiles) ? kt + 1 : kt;   // clamped lookahead (in-bounds)
    __builtin_prefetch(Bt + (size_t)(kt + 2) * 512 + lane * 16, 0, 1);
    v16h bnext = *(const v16h*)(Bt + (size_t)ktn * 512 + lane * 16);
    #pragma unroll
    for (int m = 0; m < 8; ++m) {
      v16h anext = (m < 7) ? load_a_frag(Ab, EE, (m + 1) * 16, kt * 32, lane)
                           : load_a_frag(Ab, EE, 0, ktn * 32, lane);
      acc[m] = wmma_f16(acur, bcur, acc[m]);
      acur = anext;
    }
    bcur = bnext;
  }

  int colLocal = nt * 16 + (lane & 15);
  int col      = c_coloff[br] + colLocal;           // always < 1720
  float bias   = biasCat[col];

  float mx = 0.f;                                    // relu>=0 -> 0 is identity
  #pragma unroll
  for (int m = 0; m < 8; ++m) {
    // D layout: lane<16 holds rows m*16+0..7, lane>=16 rows m*16+8..15
    int mbase = m * 16 + ((lane >> 4) << 3);
    #pragma unroll
    for (int r = 0; r < 8; ++r) {
      int t = mbase + r;
      float v = acc[m][r] + bias;
      v = v > 0.f ? v : 0.f;
      if (t < sout) mx = fmaxf(mx, v);
    }
  }
  mx = fmaxf(mx, __shfl_xor(mx, 16, 32));            // fold the two row-groups
  if (lane < 16 && colLocal < c_n[br])
    pred[(size_t)b * FTOT + col] = mx;
}

// ---- kernel 4: pred f32 -> f16, K-padded to 1728 --------------------------
__global__ void pred_to_f16(const float* __restrict__ pred, _Float16* __restrict__ pf) {
  size_t idx = (size_t)blockIdx.x * blockDim.x + threadIdx.x;
  if (idx >= (size_t)BB * HW_KPAD) return;
  int c = (int)(idx % HW_KPAD);
  size_t r = idx / HW_KPAD;
  pf[idx] = (c < FTOT) ? (_Float16)pred[r * FTOT + c] : (_Float16)0.f;
}

// ---- kernel 5: highway GEMM  h = pred @ hw_w + hw_b -----------------------
// grid = (16 M-panels of 128 rows, 108 N-tiles); 1 wave per (panel, N-tile),
// 8 M-tile accumulators, 54 K-steps, B reused across the panel, pipelined.
__global__ void __launch_bounds__(32)
highway_gemm(const _Float16* __restrict__ Af16, const _Float16* __restrict__ hwpack,
             const float* __restrict__ hw_b, float* __restrict__ h) {
  int mpanel = blockIdx.x;          // 0..15, rows mpanel*128 .. +127
  int ntt    = blockIdx.y;          // 0..107
  int lane   = threadIdx.x;
  const _Float16* Bt = hwpack + (size_t)ntt * HW_KT * 512;
  const _Float16* Ap = Af16 + (size_t)mpanel * 128 * HW_KPAD;

  v8f acc[8];
  #pragma unroll
  for (int m = 0; m < 8; ++m) acc[m] = (v8f){};

  v16h bcur = *(const v16h*)(Bt + lane * 16);
  v16h acur = load_a_frag(Ap, HW_KPAD, 0, 0, lane);

  for (int kt = 0; kt < HW_KT; ++kt) {
    int ktn = (kt + 1 < HW_KT) ? kt + 1 : kt;     // clamped lookahead (in-bounds)
    __builtin_prefetch(Bt + (size_t)(kt + 2) * 512 + lane * 16, 0, 1);
    v16h bnext = *(const v16h*)(Bt + (size_t)ktn * 512 + lane * 16);
    #pragma unroll
    for (int m = 0; m < 8; ++m) {
      v16h anext = (m < 7) ? load_a_frag(Ap, HW_KPAD, (m + 1) * 16, kt * 32, lane)
                           : load_a_frag(Ap, HW_KPAD, 0, ktn * 32, lane);
      acc[m] = wmma_f16(acur, bcur, acc[m]);
      acur = anext;
    }
    bcur = bnext;
  }

  int col = ntt * 16 + (lane & 15);
  if (col < FTOT) {
    float hb = hw_b[col];
    #pragma unroll
    for (int m = 0; m < 8; ++m) {
      int mbase = mpanel * 128 + m * 16 + ((lane >> 4) << 3);
      #pragma unroll
      for (int r = 0; r < 8; ++r)
        h[(size_t)(mbase + r) * FTOT + col] = acc[m][r] + hb;
    }
  }
}

// ---- kernel 6: highway blend + 1720x2 head + log_softmax ------------------
__global__ void __launch_bounds__(32)
final_out(const float* __restrict__ h, const float* __restrict__ pred,
          const float* __restrict__ out_w, const float* __restrict__ out_b,
          float* __restrict__ out) {
  int b = blockIdx.x;
  int lane = threadIdx.x;
  float a0 = 0.f, a1 = 0.f;
  for (int c = lane; c < FTOT; c += 32) {
    float hv = h[(size_t)b * FTOT + c];
    float g  = 1.f / (1.f + expf(-hv));
    float rl = hv > 0.f ? hv : 0.f;
    float p2 = g * rl + (1.f - g) * pred[(size_t)b * FTOT + c];
    a0 += p2 * out_w[2 * c + 0];
    a1 += p2 * out_w[2 * c + 1];
  }
  #pragma unroll
  for (int o = 16; o > 0; o >>= 1) {
    a0 += __shfl_xor(a0, o, 32);
    a1 += __shfl_xor(a1, o, 32);
  }
  if (lane == 0) {
    float z0 = a0 + out_b[0], z1 = a1 + out_b[1];
    float mm = fmaxf(z0, z1);
    float l  = logf(expf(z0 - mm) + expf(z1 - mm));
    out[2 * b + 0] = z0 - mm - l;
    out[2 * b + 1] = z1 - mm - l;
  }
}

// ---------------------------------------------------------------------------
extern "C" void kernel_launch(void* const* d_in, const int* in_sizes, int n_in,
                              void* d_out, int out_size, void* d_ws, size_t ws_size,
                              hipStream_t stream) {
  (void)in_sizes; (void)n_in; (void)out_size; (void)ws_size;

  const int*   xt    = (const int*)d_in[0];
  const int*   xs    = (const int*)d_in[1];
  const float* temb  = (const float*)d_in[2];
  const float* semb  = (const float*)d_in[3];
  // d_in[4..15]  = conv_ws, d_in[16..27] = conv_bs
  const float* hw_w  = (const float*)d_in[28];
  const float* hw_b  = (const float*)d_in[29];
  const float* out_w = (const float*)d_in[30];
  const float* out_b = (const float*)d_in[31];
  float* out = (float*)d_out;

  // ---- workspace layout (all 256B aligned) ----
  char* ws = (char*)d_ws;
  const size_t OFF_EMB    = 0;                                          // 52,428,800
  const size_t OFF_WPACK  = OFF_EMB    + (size_t)BB * SP * EE * 2;      //  2,125,824
  const size_t OFF_HWPACK = OFF_WPACK  + (size_t)1062912 * 2;           //  5,971,968
  const size_t OFF_BIAS   = OFF_HWPACK + (size_t)HW_KPAD * HW_KPAD * 2; //      6,912
  const size_t OFF_PRED   = OFF_BIAS   + 6912;                          // 14,090,240
  const size_t OFF_PF16   = OFF_PRED   + (size_t)BB * FTOT * 4;         //  7,077,888
  const size_t OFF_H      = OFF_PF16   + (size_t)BB * HW_KPAD * 2;      // 14,090,240

  _Float16* emb     = (_Float16*)(ws + OFF_EMB);
  _Float16* wpack   = (_Float16*)(ws + OFF_WPACK);
  _Float16* hwpack  = (_Float16*)(ws + OFF_HWPACK);
  float*    biasCat = (float*)   (ws + OFF_BIAS);
  float*    pred    = (float*)   (ws + OFF_PRED);
  _Float16* pf16    = (_Float16*)(ws + OFF_PF16);
  float*    hbuf    = (float*)   (ws + OFF_H);

  // host-side branch metadata
  static const int h_f[NBR]      = {1,2,3,4,5,6,7,8,9,10,15,20};
  static const int h_n[NBR]      = {100,200,200,200,200,100,100,100,100,100,160,160};
  static const int h_ktiles[NBR] = {3,5,8,10,13,15,18,20,23,25,38,50};
  static const int h_ntiles[NBR] = {7,13,13,13,13,7,7,7,7,7,10,10};
  static const int h_woff[NBR]   = {0,10752,44032,97280,163840,250368,
                                    304128,368640,440320,522752,612352,806912};
  static const int h_coloff[NBR] = {0,100,300,500,700,900,1000,1100,1200,1300,1400,1560};

  // 1) embedding gather -> f16
  {
    size_t tot = (size_t)BB * SP * EE;
    embed_gather<<<(unsigned)((tot + 255) / 256), 256, 0, stream>>>(xt, xs, temb, semb, emb);
  }

  // 2) pack conv weights + biases; pack highway weight
  for (int i = 0; i < NBR; ++i) {
    int total = h_ntiles[i] * h_ktiles[i] * 512;
    pack_b_f16<<<(total + 255) / 256, 256, 0, stream>>>(
        (const float*)d_in[4 + i], wpack + h_woff[i],
        h_f[i] * EE, h_n[i], h_ktiles[i], h_ntiles[i]);
    copy_bias<<<(h_n[i] + 63) / 64, 64, 0, stream>>>(
        (const float*)d_in[16 + i], biasCat + h_coloff[i], h_n[i]);
  }
  {
    int total = HW_NT * HW_KT * 512;
    pack_b_f16<<<(total + 255) / 256, 256, 0, stream>>>(hw_w, hwpack,
                                                        FTOT, FTOT, HW_KT, HW_NT);
  }

  // 3) conv GEMM + pool  (2048 x 114 waves)
  {
    dim3 g(BB, 114);
    conv_gemm_pool<<<g, 32, 0, stream>>>(emb, wpack, biasCat, pred);
  }

  // 4) pred -> f16 (padded)
  {
    size_t tot = (size_t)BB * HW_KPAD;
    pred_to_f16<<<(unsigned)((tot + 255) / 256), 256, 0, stream>>>(pred, pf16);
  }

  // 5) highway GEMM (16 x 108 waves, 128-row M-panels)
  {
    dim3 g(BB / 128, HW_NT);
    highway_gemm<<<g, 32, 0, stream>>>(pf16, hwpack, hw_b, hbuf);
  }

  // 6) blend + head + log_softmax
  final_out<<<BB, 32, 0, stream>>>(hbuf, pred, out_w, out_b, out);
}
